// PointHop_61392262529219
// MI455X (gfx1250) — matrix operand order repaired and checked by reading, not compiled, not run on hardware
//
#include <hip/hip_runtime.h>
#include <stdint.h>

typedef __attribute__((ext_vector_type(16))) _Float16 v16h;
typedef __attribute__((ext_vector_type(8)))  float    v8f;
typedef __attribute__((ext_vector_type(2)))  _Float16 h2;

#define Bn 32
#define Nn 8192
#define Kn 64
#define GROUPS (Bn * Nn)
#define CH_STRIDE ((size_t)Nn * Kn)
#define WAVE_LDS 1152   // 8 col regions * 128B + 128B idx

union ABu { uint32_t u[8]; v16h v; };
union HU  { h2 h; uint32_t u; };

__device__ __forceinline__ uint32_t pkh2(float a, float b) {
  union { _Float16 h[2]; uint32_t u; } t;
  t.h[0] = (_Float16)a; t.h[1] = (_Float16)b;
  return t.u;
}

// compiler-only ordering for per-wave-private LDS (hardware keeps same-wave
// LDS ops in order; no workgroup barrier needed)
__device__ __forceinline__ void wave_lds_fence() {
  __builtin_amdgcn_fence(__ATOMIC_RELEASE, "wavefront");
  __builtin_amdgcn_wave_barrier();
  __builtin_amdgcn_fence(__ATOMIC_ACQUIRE, "wavefront");
}

__global__ __launch_bounds__(256) void pointhop_wmma_kernel(
    const float* __restrict__ gx,   // [B,3,N,K]
    const float* __restrict__ nx,   // [B,N,3]
    float* __restrict__ out) {      // [B,N,30]
  // per-wave LDS: regions 0..5 = f16 channels (x,y,z,x2,y2,z2),
  // region 6 = ones (counts col), region 7 = zeros (unused cols), then f16 ids
  __shared__ __align__(16) unsigned char smem[8 * WAVE_LDS];
  const int lane = threadIdx.x & 31;
  const int wib  = threadIdx.x >> 5;
  const int wid  = blockIdx.x * 8 + wib;
  const int nw   = gridDim.x * 8;

  unsigned char* chB  = smem + wib * WAVE_LDS;
  unsigned char* idxB = chB + 1024;
  uint32_t* chW  = (uint32_t*)chB;
  uint32_t* idxW = (uint32_t*)idxB;

  const uint32_t m   = lane & 15;              // A row id / B-D column id
  const int      kb  = (lane < 16) ? 0 : 8;    // B K-base within 32-chunk
  const int      ab  = (lane < 16) ? 0 : 8;    // A K-base within 32-chunk
  const uint32_t csel = (m < 7u) ? m : 7u;     // B column -> LDS region
  // rows 8..15: never-matching target -> indicator always 0 (no cndmask)
  const float    mtgt = (m < 8u) ? (float)m : 64.0f;
  const h2 mh2  = { (_Float16)mtgt, (_Float16)mtgt };
  const h2 one2 = { (_Float16)1.0f, (_Float16)1.0f };
  const h2 zer2 = { (_Float16)0.0f, (_Float16)0.0f };

  // constant column regions, written once (wave-private, persist across iters)
  chW[6 * 32 + lane] = 0x3C003C00u;   // ones
  chW[7 * 32 + lane] = 0u;            // zeros

  for (int g = wid; g < GROUPS; g += nw) {
    const int b = g >> 13, n = g & (Nn - 1);
    const float* p0 = gx + ((size_t)(b * 3) * Nn + n) * Kn + 2 * lane;

    // prefetch next group's cachelines (global_prefetch_b8)
    const int gn = g + nw;
    if (gn < GROUPS) {
      const int b2 = gn >> 13, n2 = gn & (Nn - 1);
      const float* q0 = gx + ((size_t)(b2 * 3) * Nn + n2) * Kn + 2 * lane;
      __builtin_prefetch(q0, 0, 1);
      __builtin_prefetch(q0 + CH_STRIDE, 0, 1);
      __builtin_prefetch(q0 + 2 * CH_STRIDE, 0, 1);
    }

    // coalesced stream: 256B per wave per channel, each byte read exactly once
    float2 vx = *(const float2*)(p0);
    float2 vy = *(const float2*)(p0 + CH_STRIDE);
    float2 vz = *(const float2*)(p0 + 2 * CH_STRIDE);

    // octant ids for this lane's two points, staged as exact f16 pair
    uint32_t i0 = (vx.x > 0.f ? 4u : 0u) | (vy.x > 0.f ? 2u : 0u) | (vz.x > 0.f ? 1u : 0u);
    uint32_t i1 = (vx.y > 0.f ? 4u : 0u) | (vy.y > 0.f ? 2u : 0u) | (vz.y > 0.f ? 1u : 0u);
    idxW[lane] = pkh2((float)i0, (float)i1);

    // f16 staging; squares via one v_pk_mul_f16 on the packed pair
    HU hx, hy, hz, sx, sy, sz;
    hx.u = pkh2(vx.x, vx.y);  sx.h = hx.h * hx.h;
    hy.u = pkh2(vy.x, vy.y);  sy.h = hy.h * hy.h;
    hz.u = pkh2(vz.x, vz.y);  sz.h = hz.h * hz.h;
    chW[0 * 32 + lane] = hx.u;
    chW[1 * 32 + lane] = hy.u;
    chW[2 * 32 + lane] = hz.u;
    chW[3 * 32 + lane] = sx.u;
    chW[4 * 32 + lane] = sy.u;
    chW[5 * 32 + lane] = sz.u;

    wave_lds_fence();

    v8f acc = {};
#pragma unroll
    for (int t = 0; t < 2; ++t) {
      // ---- A fragment: 16x32 f16 one-hot via packed-f16 arithmetic:
      // ind = max(1 - (idx - m)^2, 0); rows>=8 use m=64 -> always 0
      const int a1 = t * 32 + ab;   // this lane's K base
      const uint4 qa0 = *(const uint4*)(idxB + a1 * 2);
      const uint4 qa1 = *(const uint4*)(idxB + a1 * 2 + 32);
      uint32_t aw[8] = {qa0.x, qa0.y, qa0.z, qa0.w, qa1.x, qa1.y, qa1.z, qa1.w};
      ABu A;
#pragma unroll
      for (int i = 0; i < 8; ++i) {
        HU v; v.u = aw[i];
        h2 d = v.h - mh2;                          // v_pk_add neg
        h2 r = one2 - d * d;                       // v_pk_fma
        r = __builtin_elementwise_max(r, zer2);    // v_pk_max_num_f16
        HU o; o.h = r;
        A.u[i] = o.u;
      }

      // ---- B fragment: 32x16 f16, branchless region lookup ----
      const uint4* pB = (const uint4*)(chB + csel * 128 + (size_t)(t * 32 + kb) * 2);
      uint4 q0 = pB[0], q1 = pB[2];
      ABu Bm;
      Bm.u[0] = q0.x; Bm.u[1] = q0.y; Bm.u[2] = q0.z; Bm.u[3] = q0.w;
      Bm.u[4] = q1.x; Bm.u[5] = q1.y; Bm.u[6] = q1.z; Bm.u[7] = q1.w;

      acc = __builtin_amdgcn_wmma_f32_16x16x32_f16(
          false, A.v, false, Bm.v, (short)0, acc, false, false);
    }

    // ---- extract ----
    // lanes 0..15 hold D[e][col=lane]; bins partition K, so the column sum
    // over e gives: cols 0..2 -> Σx_c, cols 3..5 -> Σx²_c, col 6 -> 64
    float s = ((acc[0] + acc[1]) + (acc[2] + acc[3])) +
              ((acc[4] + acc[5]) + (acc[6] + acc[7]));
    // per-bin counts D[e][6]: uniform broadcast via v_readlane
    float cnt[8];
#pragma unroll
    for (int e = 0; e < 8; ++e)
      cnt[e] = __int_as_float(__builtin_amdgcn_readlane(__float_as_int(acc[e]), 6));
    // bring Σx_c (lanes 0..2) to the Σx²_c lanes (3..5)
    float sumv = __shfl(s, lane - 3, 32);

    const size_t ob = (size_t)g * 30;
    if (lane >= 3 && lane < 6) {
      // std_c = sqrt((Σx² - (Σx)²/K) / (K-1)), channel = lane-3
      float var = (s - sumv * sumv * (1.0f / 64.0f)) * (1.0f / 63.0f);
      out[ob + (lane - 3)] = __builtin_amdgcn_sqrtf(fmaxf(var, 0.0f));
    }
    if (lane < 3) {
      out[ob + 3 + lane] = nx[(size_t)g * 3 + lane];               // center
#pragma unroll
      for (int e = 0; e < 8; ++e) {                                // scatter-mean
        float r = __builtin_amdgcn_rcpf(fmaxf(cnt[e], 1.0f));      // exact ints
        out[ob + 6 + e * 3 + lane] = acc[e] * r;
      }
    }
    wave_lds_fence();
  }
}

extern "C" void kernel_launch(void* const* d_in, const int* in_sizes, int n_in,
                              void* d_out, int out_size, void* d_ws, size_t ws_size,
                              hipStream_t stream) {
  const float* gx = (const float*)d_in[0];   // group_xyz [32,3,8192,64] f32
  const float* nx = (const float*)d_in[1];   // new_xyz   [32,8192,3]   f32
  float* out = (float*)d_out;                // [32,8192,30] f32
  (void)in_sizes; (void)n_in; (void)out_size; (void)d_ws; (void)ws_size;
  // 1024 blocks * 8 waves = 8192 waves; 262144 groups -> exactly 32 iters/wave
  hipLaunchKernelGGL(pointhop_wmma_kernel, dim3(1024), dim3(256), 0, stream,
                     gx, nx, out);
}
